// SMPL_21801253994771
// MI455X (gfx1250) — matrix operand order repaired and checked by reading, not compiled, not run on hardware
//
#include <hip/hip_runtime.h>
#include <hip/hip_bf16.h>

#define NB      512
#define NV      6890
#define NJ      24
#define M_ROWS  20670          // NV*3
#define M_PAD   20672          // padded to multiple of 16
#define KDIM    207
#define KPAD    224            // 7 chunks of K=32

// ---------------- workspace layout (bytes) ----------------
#define OFF_VSH   ((size_t)0)                                   // float [NB][M_ROWS]   42,332,160
#define OFF_PD    (OFF_VSH  + (size_t)NB*M_ROWS*4)              // bf16  [M_PAD][KPAD]   9,261,056
#define OFF_LROT  (OFF_PD   + (size_t)M_PAD*KPAD*2)             // bf16  [NB][KPAD]        229,376
#define OFF_R     (OFF_LROT + (size_t)NB*KPAD*2)                // float [NB][NJ][9]       442,368
#define OFF_J     (OFF_R    + (size_t)NB*NJ*9*4)                // float [NB][NJ][3]       147,456
#define OFF_G     (OFF_J    + (size_t)NB*NJ*3*4)                // float [NB][NJ][12]      589,824

typedef __attribute__((ext_vector_type(16))) __bf16 v16bf;
typedef __attribute__((ext_vector_type(8)))  float  v8f;

union Frag16 { v16bf v; uint4 q[2]; };

static __device__ __forceinline__ unsigned short f32_to_bf16(float f) {
    unsigned int u = __float_as_uint(f);
    unsigned int r = (u + 0x7FFFu + ((u >> 16) & 1u)) >> 16;   // RNE
    return (unsigned short)r;
}

// ---------- 1) posedirs fp32 -> bf16, padded [M_PAD][KPAD] ----------
__global__ __launch_bounds__(256) void convert_pd_kernel(const float* __restrict__ pd,
                                                         unsigned short* __restrict__ out) {
    int i = blockIdx.x * 256 + threadIdx.x;
    if (i >= M_PAD * KPAD) return;
    int row = i / KPAD, k = i % KPAD;
    float v = (row < M_ROWS && k < KDIM) ? pd[(size_t)row * KDIM + k] : 0.0f;
    out[i] = f32_to_bf16(v);
}

// ---------- 2) v_shaped = shapedirs @ beta + v_template ----------
__global__ __launch_bounds__(256) void vshaped_kernel(const float* __restrict__ shapedirs,
                                                      const float* __restrict__ beta,
                                                      const float* __restrict__ vt,
                                                      float* __restrict__ vsh) {
    __shared__ float sb[10];
    int b = blockIdx.y, tid = threadIdx.x;
    if (tid < 10) sb[tid] = beta[b * 10 + tid];
    __syncthreads();
    int v = blockIdx.x * 256 + tid;
    if (v >= NV) return;
    #pragma unroll
    for (int c = 0; c < 3; ++c) {
        const float* sd = shapedirs + (size_t)(v * 3 + c) * 10;
        float s = vt[v * 3 + c];
        #pragma unroll
        for (int t = 0; t < 10; ++t) s += sd[t] * sb[t];
        vsh[(size_t)b * M_ROWS + v * 3 + c] = s;
    }
}

// ---------- 3) Rodrigues + lrotmin (bf16) ----------
__global__ __launch_bounds__(256) void rodrigues_kernel(const float* __restrict__ pose,
                                                        float* __restrict__ Rws,
                                                        unsigned short* __restrict__ lrot) {
    int id = blockIdx.x * 256 + threadIdx.x;
    if (id >= NB * NJ) return;
    int b = id / NJ, j = id % NJ;
    float rx = pose[b * 72 + j * 3 + 0];
    float ry = pose[b * 72 + j * 3 + 1];
    float rz = pose[b * 72 + j * 3 + 2];
    float theta = sqrtf(rx * rx + ry * ry + rz * rz) + 1e-8f;
    float inv = 1.0f / theta;
    float ax = rx * inv, ay = ry * inv, az = rz * inv;
    float c = cosf(theta), s = sinf(theta), o = 1.0f - c;
    float R[9];
    R[0] = c + o * ax * ax;      R[1] = o * ax * ay - s * az; R[2] = o * ax * az + s * ay;
    R[3] = o * ay * ax + s * az; R[4] = c + o * ay * ay;      R[5] = o * ay * az - s * ax;
    R[6] = o * az * ax - s * ay; R[7] = o * az * ay + s * ax; R[8] = c + o * az * az;
    float* ro = Rws + (size_t)(b * NJ + j) * 9;
    #pragma unroll
    for (int k = 0; k < 9; ++k) ro[k] = R[k];
    if (j > 0) {
        unsigned short* lp = lrot + (size_t)b * KPAD + (j - 1) * 9;
        #pragma unroll
        for (int k = 0; k < 9; ++k) {
            float d = R[k] - ((k == 0 || k == 4 || k == 8) ? 1.0f : 0.0f);
            lp[k] = f32_to_bf16(d);
        }
    } else {
        unsigned short* lp = lrot + (size_t)b * KPAD;
        for (int k = KDIM; k < KPAD; ++k) lp[k] = 0;
    }
}

// ---------- 4) J = J_regressor @ v_shaped ----------
__global__ __launch_bounds__(256) void jreg_kernel(const float* __restrict__ Jreg,
                                                   const float* __restrict__ vsh,
                                                   float* __restrict__ Jout) {
    __shared__ float s0[256], s1[256], s2[256];
    int j = blockIdx.x, b = blockIdx.y, tid = threadIdx.x;
    float a0 = 0.f, a1 = 0.f, a2 = 0.f;
    for (int v = tid; v < NV; v += 256) {
        float w = Jreg[(size_t)j * NV + v];
        const float* p = vsh + (size_t)b * M_ROWS + v * 3;
        a0 += w * p[0]; a1 += w * p[1]; a2 += w * p[2];
    }
    s0[tid] = a0; s1[tid] = a1; s2[tid] = a2;
    __syncthreads();
    for (int off = 128; off > 0; off >>= 1) {
        if (tid < off) { s0[tid] += s0[tid + off]; s1[tid] += s1[tid + off]; s2[tid] += s2[tid + off]; }
        __syncthreads();
    }
    if (tid == 0) {
        float* o = Jout + (size_t)(b * NJ + j) * 3;
        o[0] = s0[0]; o[1] = s1[0]; o[2] = s2[0];
    }
}

// ---------- 5) kinematic chain + rest subtraction (1 thread / batch) ----------
__global__ __launch_bounds__(256) void chain_kernel(const float* __restrict__ Rws,
                                                    const float* __restrict__ Jws,
                                                    float* __restrict__ Gws) {
    int b = blockIdx.x * 256 + threadIdx.x;
    if (b >= NB) return;
    const int par[23] = {0,0,0,1,2,3,4,5,6,7,8,9,9,9,12,13,14,16,17,18,19,20,21};
    float G[NJ][12];
    const float* Rb = Rws + (size_t)b * NJ * 9;
    const float* Jb = Jws + (size_t)b * NJ * 3;
    // root
    G[0][0]=Rb[0]; G[0][1]=Rb[1]; G[0][2]=Rb[2];  G[0][3]=Jb[0];
    G[0][4]=Rb[3]; G[0][5]=Rb[4]; G[0][6]=Rb[5];  G[0][7]=Jb[1];
    G[0][8]=Rb[6]; G[0][9]=Rb[7]; G[0][10]=Rb[8]; G[0][11]=Jb[2];
    for (int i = 1; i < NJ; ++i) {
        int p = par[i - 1];
        const float* Rl = Rb + i * 9;
        float t0 = Jb[i*3+0] - Jb[p*3+0];
        float t1 = Jb[i*3+1] - Jb[p*3+1];
        float t2 = Jb[i*3+2] - Jb[p*3+2];
        const float* Gp = G[p];
        #pragma unroll
        for (int r = 0; r < 3; ++r) {
            float g0 = Gp[r*4+0], g1 = Gp[r*4+1], g2 = Gp[r*4+2], g3 = Gp[r*4+3];
            G[i][r*4+0] = g0*Rl[0] + g1*Rl[3] + g2*Rl[6];
            G[i][r*4+1] = g0*Rl[1] + g1*Rl[4] + g2*Rl[7];
            G[i][r*4+2] = g0*Rl[2] + g1*Rl[5] + g2*Rl[8];
            G[i][r*4+3] = g0*t0 + g1*t1 + g2*t2 + g3;
        }
    }
    float* Go = Gws + (size_t)b * NJ * 12;
    for (int j = 0; j < NJ; ++j) {
        float jx = Jb[j*3], jy = Jb[j*3+1], jz = Jb[j*3+2];
        #pragma unroll
        for (int r = 0; r < 3; ++r) {
            float g0 = G[j][r*4], g1 = G[j][r*4+1], g2 = G[j][r*4+2];
            Go[j*12+r*4+0] = g0;
            Go[j*12+r*4+1] = g1;
            Go[j*12+r*4+2] = g2;
            Go[j*12+r*4+3] = G[j][r*4+3] - (g0*jx + g1*jy + g2*jz);
        }
    }
}

// ---------- 6) v_posed = posedirs @ lrotmin + v_shaped  (bf16 WMMA) ----------
// A = lrotmin [NB x KPAD]   (M = batch)
// B = posedirs^T, fed from row-major pd_bf16 [M_PAD x KPAD] (N = v*3+c rows)
// D[m,n] -> out[b*M_ROWS + g]  (lanes 0-15 write 16 consecutive floats)
__global__ __launch_bounds__(256) void posed_wmma_kernel(const unsigned short* __restrict__ lrot,
                                                         const unsigned short* __restrict__ pd,
                                                         const float* __restrict__ vsh,
                                                         float* __restrict__ out) {
    const int wave = threadIdx.x >> 5;
    const int lane = threadIdx.x & 31;
    const int tile = blockIdx.x * 8 + wave;     // 41344 tiles total = 5168 * 8
    const int tM = tile & 31;                   // batch tile  (512/16 = 32)
    const int tN = tile >> 5;                   // row tile    (20672/16 = 1292)
    const int h = lane >> 4;
    const int l = lane & 15;

    const unsigned short* aptr = lrot + (size_t)(tM * 16 + l) * KPAD;   // batch row
    const unsigned short* bptr = pd   + (size_t)(tN * 16 + l) * KPAD;   // posedirs row

    v8f acc = {0.f, 0.f, 0.f, 0.f, 0.f, 0.f, 0.f, 0.f};
    #pragma unroll
    for (int kc = 0; kc < 7; ++kc) {
        const int k0 = kc * 32;
        Frag16 A, Bm;
        // A 16x32 bf16: lane holds K = {h*8..h*8+7} and {16+h*8..16+h*8+7}
        A.q[0]  = *(const uint4*)(aptr + k0 + h * 8);
        A.q[1]  = *(const uint4*)(aptr + k0 + 16 + h * 8);
        // B 32x16 bf16: lane holds K = {h*8..h*8+7} and {16+h*8..16+h*8+7}
        Bm.q[0] = *(const uint4*)(bptr + k0 + h * 8);
        Bm.q[1] = *(const uint4*)(bptr + k0 + 16 + h * 8);
        acc = __builtin_amdgcn_wmma_f32_16x16x32_bf16(false, A.v, false, Bm.v,
                                                      (short)0, acc, false, false);
    }
    const int g = tN * 16 + l;                  // output row (v*3+c)
    if (g < M_ROWS) {
        #pragma unroll
        for (int r = 0; r < 8; ++r) {
            const int bb = tM * 16 + 8 * h + r; // batch
            const size_t idx = (size_t)bb * M_ROWS + g;
            out[idx] = acc[r] + vsh[idx];
        }
    }
}

// ---------- 7) skinning: v = (sum_j w[v,j] G[b,j]) * [v_posed,1] ----------
__global__ __launch_bounds__(256) void skin_kernel(const float* __restrict__ Gws,
                                                   const float* __restrict__ weights,
                                                   float* __restrict__ out) {
    __shared__ float sG[NJ * 12];
    int b = blockIdx.y, tid = threadIdx.x;
    for (int i = tid; i < NJ * 12; i += 256) sG[i] = Gws[(size_t)b * NJ * 12 + i];
    __syncthreads();
    int v = blockIdx.x * 256 + tid;
    if (v >= NV) return;
    size_t base = (size_t)b * M_ROWS + v * 3;
    float p0 = out[base], p1 = out[base + 1], p2 = out[base + 2];  // v_posed (in place)
    float a0 = 0.f, a1 = 0.f, a2 = 0.f;
    const float* w = weights + (size_t)v * NJ;
    #pragma unroll 4
    for (int j = 0; j < NJ; ++j) {
        float wj = w[j];
        const float* g = sG + j * 12;
        a0 += wj * (g[0]*p0 + g[1]*p1 + g[2]*p2  + g[3]);
        a1 += wj * (g[4]*p0 + g[5]*p1 + g[6]*p2  + g[7]);
        a2 += wj * (g[8]*p0 + g[9]*p1 + g[10]*p2 + g[11]);
    }
    out[base] = a0; out[base + 1] = a1; out[base + 2] = a2;
}

extern "C" void kernel_launch(void* const* d_in, const int* in_sizes, int n_in,
                              void* d_out, int out_size, void* d_ws, size_t ws_size,
                              hipStream_t stream) {
    const float* pose      = (const float*)d_in[0];
    const float* beta      = (const float*)d_in[1];
    const float* vtemplate = (const float*)d_in[2];
    const float* shapedirs = (const float*)d_in[3];
    const float* posedirs  = (const float*)d_in[4];
    const float* Jreg      = (const float*)d_in[5];
    const float* weights   = (const float*)d_in[6];
    float* out = (float*)d_out;

    char* ws = (char*)d_ws;
    float*          vsh  = (float*)(ws + OFF_VSH);
    unsigned short* pdb  = (unsigned short*)(ws + OFF_PD);
    unsigned short* lrot = (unsigned short*)(ws + OFF_LROT);
    float*          Rws  = (float*)(ws + OFF_R);
    float*          Jws  = (float*)(ws + OFF_J);
    float*          Gws  = (float*)(ws + OFF_G);

    convert_pd_kernel<<<(M_PAD * KPAD + 255) / 256, 256, 0, stream>>>(posedirs, pdb);
    vshaped_kernel<<<dim3((NV + 255) / 256, NB), 256, 0, stream>>>(shapedirs, beta, vtemplate, vsh);
    rodrigues_kernel<<<(NB * NJ + 255) / 256, 256, 0, stream>>>(pose, Rws, lrot);
    jreg_kernel<<<dim3(NJ, NB), 256, 0, stream>>>(Jreg, vsh, Jws);
    chain_kernel<<<(NB + 255) / 256, 256, 0, stream>>>(Rws, Jws, Gws);
    posed_wmma_kernel<<<(32 * (M_PAD / 16)) / 8, 256, 0, stream>>>(lrot, pdb, vsh, out);
    skin_kernel<<<dim3((NV + 255) / 256, NB), 256, 0, stream>>>(Gws, weights, out);
}
